// PointNetMSGInputPoint_42236708389449
// MI455X (gfx1250) — matrix operand order repaired and checked by reading, not compiled, not run on hardware
//
#include <hip/hip_runtime.h>
#include <hip/hip_bf16.h>

// ---------------------------------------------------------------------------
// PointNet++ MSG layer for MI455X (gfx1250).
//  - wave32 ball query (ballot + prefix-popcount append)
//  - bf16 WMMA GEMMs, register-blocked over N (A read exactly once),
//    B k-tile double-buffered in LDS (b128 DS traffic, 1 barrier/iter)
//  - training-mode BN stats via f32 atomics in the GEMM epilogue
//  - conv bias dropped: it cancels exactly under training-mode BN
// ---------------------------------------------------------------------------

#define EPSV 1e-5f

typedef __attribute__((ext_vector_type(16))) __bf16 v16bf;
typedef __attribute__((ext_vector_type(8)))  float  v8f;

union FragAB {
    v16bf v;
    unsigned int d[8];
    uint4 q[2];
};

__device__ __forceinline__ unsigned short f2bf(float f) {
    unsigned int u = __builtin_bit_cast(unsigned int, f);
    unsigned int r = (u + 0x7fffu + ((u >> 16) & 1u)) >> 16;   // RNE
    return (unsigned short)r;
}
__device__ __forceinline__ float bf2f(unsigned short h) {
    unsigned int u = ((unsigned int)h) << 16;
    return __builtin_bit_cast(float, u);
}
__device__ __forceinline__ unsigned int pack2(float lo, float hi) {
    return (unsigned int)f2bf(lo) | ((unsigned int)f2bf(hi) << 16);
}

// ---------------------------------------------------------------------------
// Ball query: one wave32 per centroid.
// ---------------------------------------------------------------------------
__global__ void ball_query_kernel(const float* __restrict__ xyz,
                                  const float* __restrict__ new_xyz,
                                  int* __restrict__ idx,
                                  int Bn, int N, int S, int K, float r2) {
    int wave = (blockIdx.x * blockDim.x + threadIdx.x) >> 5;
    int lane = threadIdx.x & 31;
    int total = Bn * S;
    if (wave >= total) return;
    int b = wave / S;
    const float* c = new_xyz + (size_t)wave * 3;
    float cx = c[0], cy = c[1], cz = c[2];
    int* out = idx + (size_t)wave * K;
    int cnt = 0, first = 0;
    for (int n0 = 0; n0 < N && cnt < K; n0 += 32) {
        int p = n0 + lane;                       // N % 32 == 0, always valid
        const float* q = xyz + ((size_t)b * N + p) * 3;
        float dx = q[0] - cx, dy = q[1] - cy, dz = q[2] - cz;
        float sq = dx * dx + dy * dy + dz * dz;
        bool pred = (sq <= r2);
        unsigned int m32 = (unsigned int)__ballot(pred);   // wave32
        if (cnt == 0 && m32) {
            int fl = __ffs(m32) - 1;
            first = __shfl(p, fl, 32);
        }
        int pos = cnt + __popc(m32 & ((1u << lane) - 1u));
        if (pred && pos < K) out[pos] = p;
        cnt += __popc(m32);
    }
    if (cnt > K) cnt = K;
    for (int j = cnt + lane; j < K; j += 32) out[j] = first;  // pad w/ first
}

// ---------------------------------------------------------------------------
// Gather (feat || xyz - centroid) -> bf16 A matrix [M, 96], dword stores
// ---------------------------------------------------------------------------
__global__ void gather_kernel(const float* __restrict__ xyz,
                              const float* __restrict__ new_xyz,
                              const float* __restrict__ feat,
                              const int* __restrict__ idx,
                              unsigned short* __restrict__ A,
                              int Bn, int N, int S, int D, int K, int Kc) {
    int M = Bn * S * K;
    for (int row = blockIdx.x * blockDim.x + threadIdx.x; row < M;
         row += gridDim.x * blockDim.x) {
        int bs = row / K;
        int b = bs / S;
        int p = idx[row];
        unsigned int* out = (unsigned int*)(A + (size_t)row * Kc);
        const float* f = feat + ((size_t)b * N + p) * D;
#pragma unroll 8
        for (int c = 0; c < D / 2; ++c) out[c] = pack2(f[2 * c], f[2 * c + 1]);
        const float* q  = xyz + ((size_t)b * N + p) * 3;
        const float* cp = new_xyz + (size_t)bs * 3;
        out[D / 2]     = pack2(q[0] - cp[0], q[1] - cp[1]);
        out[D / 2 + 1] = pack2(q[2] - cp[2], 0.f);
        for (int c = D / 2 + 2; c < Kc / 2; ++c) out[c] = 0u;
    }
}

// Convert f32 weight [cout,cin] -> zero-padded bf16 [cout,Kc]
__global__ void convw_kernel(const float* __restrict__ W,
                             unsigned short* __restrict__ Wb,
                             int cout, int cin, int Kc) {
    int total = cout * Kc;
    for (int i = blockIdx.x * blockDim.x + threadIdx.x; i < total;
         i += gridDim.x * blockDim.x) {
        int n = i / Kc, k = i - n * Kc;
        Wb[i] = (k < cin) ? f2bf(W[(size_t)n * cin + k]) : (unsigned short)0;
    }
}

__global__ void zero_kernel(float* p, int n) {
    int i = blockIdx.x * blockDim.x + threadIdx.x;
    if (i < n) p[i] = 0.f;
}

// ---------------------------------------------------------------------------
// bf16 WMMA GEMM, register-blocked over N, double-buffered B in LDS:
//   Y[M, NT*16] = A[M,Kc] x W^T   (W stored [N,Kc] row-major)
// Each wave owns 16 M-rows and ALL NT n-tiles (acc in regs, A loaded once).
// LDS row stride = 40 ushorts (80 B): rows 16-B aligned (ds_load_b128) and
// start banks {20*ml mod 64} give disjoint 4-dword ranges -> conflict-free.
// One barrier per k-tile: stage tile t+1 into buf[(t+1)&1] while computing t.
// Epilogue: bf16 Y + atomic per-channel sum / sumsq for BN stats.
// ---------------------------------------------------------------------------
template <int NT>
__global__ void gemm_bf16_kernel(const unsigned short* __restrict__ A,
                                 const unsigned short* __restrict__ W,
                                 unsigned short* __restrict__ Y,
                                 float* __restrict__ sum,
                                 float* __restrict__ sumsq,
                                 int M, int Kc) {
    constexpr int N    = NT * 16;
    constexpr int BSTR = 40;                      // LDS row stride (ushorts)
    __shared__ unsigned short Bs[2][N * BSTR];
    int waveInBlock = threadIdx.x >> 5;
    int lane = threadIdx.x & 31;
    int m0 = (blockIdx.x * 8 + waveInBlock) * 16;   // M % 128 == 0 always
    int half = lane >> 4;
    int ml   = lane & 15;
    const unsigned short* Arow = A + (size_t)(m0 + ml) * Kc;

    const int TILES = Kc >> 5;

    auto stage = [&](int t, int buf) {
        // Bs[buf][n*BSTR + kk] = W[n*Kc + t*32 + kk], kk in [0,32)
        for (int i = threadIdx.x; i < N * 16; i += 256) {
            int n = i >> 4;
            int j = i & 15;
            *(unsigned int*)&Bs[buf][n * BSTR + 2 * j] =
                *(const unsigned int*)(W + (size_t)n * Kc + (t << 5) + 2 * j);
        }
    };

    v8f acc[NT];
#pragma unroll
    for (int nt = 0; nt < NT; ++nt) acc[nt] = v8f{};

    stage(0, 0);
    __syncthreads();

    for (int t = 0; t < TILES; ++t) {
        int kt = t << 5;
        if (t + 1 < TILES) {
            __builtin_prefetch(Arow + kt + 32, 0, 1);   // next A k-tile
            stage(t + 1, (t + 1) & 1);
        }
        const unsigned short* bsc = Bs[t & 1];

        FragAB a;                                  // 16-bit A 16x32 layout:
        a.q[0] = *(const uint4*)(Arow + kt + half * 8);        // K half*8..+7
        a.q[1] = *(const uint4*)(Arow + kt + 16 + half * 8);   // K 16+half*8..
#pragma unroll
        for (int nt = 0; nt < NT; ++nt) {
            FragAB b;                              // 16-bit B 32x16 layout
            const unsigned short* brow = bsc + (nt * 16 + ml) * BSTR + half * 8;
            b.q[0] = *(const uint4*)(brow);
            b.q[1] = *(const uint4*)(brow + 16);
            acc[nt] = __builtin_amdgcn_wmma_f32_16x16x32_bf16(
                false, a.v, false, b.v, (short)0, acc[nt], false, false);
        }
        __syncthreads();       // staging of t+1 done; buf[t&1] free to rewrite
    }

#pragma unroll
    for (int nt = 0; nt < NT; ++nt) {
        float ls = 0.f, lss = 0.f;
#pragma unroll
        for (int r = 0; r < 8; ++r) {
            float y = acc[nt][r];
            int m = m0 + half * 8 + r;            // C/D layout (ISA 7.12.2)
            Y[(size_t)m * N + nt * 16 + ml] = f2bf(y);
            ls += y; lss += y * y;
        }
        atomicAdd(&sum[nt * 16 + ml], ls);
        atomicAdd(&sumsq[nt * 16 + ml], lss);
    }
}

// BN (training stats) + PReLU -> bf16 activations for next layer
__global__ void bn_prelu_kernel(const unsigned short* __restrict__ Y,
                                const float* __restrict__ sum,
                                const float* __restrict__ sumsq,
                                const float* __restrict__ g,
                                const float* __restrict__ be,
                                const float* __restrict__ al,
                                unsigned short* __restrict__ Out,
                                int M, int N) {
    size_t total = (size_t)M * N;
    float invM = 1.0f / (float)M;
    for (size_t i = blockIdx.x * (size_t)blockDim.x + threadIdx.x; i < total;
         i += (size_t)gridDim.x * blockDim.x) {
        int n = (int)(i % N);
        float mean = sum[n] * invM;
        float var  = sumsq[n] * invM - mean * mean;
        float x = (bf2f(Y[i]) - mean) * rsqrtf(var + EPSV) * g[n] + be[n];
        x = (x >= 0.f) ? x : al[n] * x;
        Out[i] = f2bf(x);
    }
}

// BN + PReLU + max over K -> concat buffer columns [coff, coff+N)
__global__ void bn_prelu_max_kernel(const unsigned short* __restrict__ Y,
                                    const float* __restrict__ sum,
                                    const float* __restrict__ sumsq,
                                    const float* __restrict__ g,
                                    const float* __restrict__ be,
                                    const float* __restrict__ al,
                                    unsigned short* __restrict__ Cat,
                                    int BS, int K, int N, int catStride, int coff) {
    int total = BS * N;
    float invM = 1.0f / (float)(BS * K);
    for (int i = blockIdx.x * blockDim.x + threadIdx.x; i < total;
         i += gridDim.x * blockDim.x) {
        int n = i % N, bs = i / N;
        float mean = sum[n] * invM;
        float istd = rsqrtf(sumsq[n] * invM - mean * mean + EPSV);
        float gg = g[n], bb = be[n], aa = al[n];
        float mx = -3.4e38f;
        const unsigned short* yp = Y + ((size_t)bs * K) * N + n;
        for (int k = 0; k < K; ++k) {
            float x = (bf2f(yp[(size_t)k * N]) - mean) * istd * gg + bb;
            x = (x >= 0.f) ? x : aa * x;
            mx = fmaxf(mx, x);
        }
        Cat[(size_t)bs * catStride + coff + n] = f2bf(mx);
    }
}

// Final BN + PReLU -> f32 output
__global__ void bn_prelu_out_kernel(const unsigned short* __restrict__ Y,
                                    const float* __restrict__ sum,
                                    const float* __restrict__ sumsq,
                                    const float* __restrict__ g,
                                    const float* __restrict__ be,
                                    const float* __restrict__ al,
                                    float* __restrict__ Out, int M, int N) {
    size_t total = (size_t)M * N;
    float invM = 1.0f / (float)M;
    for (size_t i = blockIdx.x * (size_t)blockDim.x + threadIdx.x; i < total;
         i += (size_t)gridDim.x * blockDim.x) {
        int n = (int)(i % N);
        float mean = sum[n] * invM;
        float var  = sumsq[n] * invM - mean * mean;
        float x = (bf2f(Y[i]) - mean) * rsqrtf(var + EPSV) * g[n] + be[n];
        Out[i] = (x >= 0.f) ? x : al[n] * x;
    }
}

// ---------------------------------------------------------------------------
static void launch_gemm(const unsigned short* A, const unsigned short* W,
                        unsigned short* Y, float* sum, float* sq,
                        int M, int cout, int Kc, hipStream_t s) {
    dim3 grid(M / 128);
    if (cout == 64)
        gemm_bf16_kernel<4><<<grid, 256, 0, s>>>(A, W, Y, sum, sq, M, Kc);
    else if (cout == 128)
        gemm_bf16_kernel<8><<<grid, 256, 0, s>>>(A, W, Y, sum, sq, M, Kc);
    else
        gemm_bf16_kernel<16><<<grid, 256, 0, s>>>(A, W, Y, sum, sq, M, Kc);
}

extern "C" void kernel_launch(void* const* d_in, const int* in_sizes, int n_in,
                              void* d_out, int out_size, void* d_ws, size_t ws_size,
                              hipStream_t stream) {
    (void)in_sizes; (void)n_in; (void)out_size; (void)ws_size;
    const int Bn = 4, N = 4096, S = 1024, D = 64, BS = Bn * S;

    const float* xyz  = (const float*)d_in[0];
    const float* nxyz = (const float*)d_in[1];
    const float* feat = (const float*)d_in[2];
    // params flattened: 3 blocks x 3 layers x 5 tensors (W,b,g,be,a)
    auto P = [&](int blk, int ly, int it) {
        return (const float*)d_in[3 + blk * 15 + ly * 5 + it];
    };

    char* ws = (char*)d_ws;
    size_t off = 0;
    auto alloc = [&](size_t bytes) {
        size_t o = off; off += (bytes + 255) & ~(size_t)255; return o;
    };
    float*          sumBuf = (float*)(ws + alloc(256 * sizeof(float)));
    float*          sqBuf  = (float*)(ws + alloc(256 * sizeof(float)));
    unsigned short* Wb     = (unsigned short*)(ws + alloc((size_t)256 * 384 * 2));
    int*            idxBuf = (int*)(ws + alloc((size_t)BS * 64 * sizeof(int)));
    unsigned short* Abuf   = (unsigned short*)(ws + alloc((size_t)BS * 64 * 128 * 2));
    unsigned short* Ybuf   = (unsigned short*)(ws + alloc((size_t)BS * 64 * 256 * 2));
    unsigned short* Cat    = (unsigned short*)(ws + alloc((size_t)BS * 384 * 2));

    const float radii[2] = {0.2f, 0.4f};
    const int   Ks[2]    = {32, 64};
    const int cins [3][3] = {{ 67,  64,  64}, { 67, 128, 128}, {384, 128, 128}};
    const int couts[3][3] = {{ 64,  64, 128}, {128, 128, 256}, {128, 128, 256}};
    const int Kcs  [3][3] = {{ 96,  64,  64}, { 96, 128, 128}, {384, 128, 128}};

    int coff = 0;
    for (int sc = 0; sc < 2; ++sc) {
        int K = Ks[sc];
        float r2 = radii[sc] * radii[sc];
        int M = BS * K;
        ball_query_kernel<<<BS / 8, 256, 0, stream>>>(xyz, nxyz, idxBuf,
                                                      Bn, N, S, K, r2);
        gather_kernel<<<1024, 256, 0, stream>>>(xyz, nxyz, feat, idxBuf, Abuf,
                                                Bn, N, S, D, K, 96);
        for (int ly = 0; ly < 3; ++ly) {
            int cin = cins[sc][ly], cout = couts[sc][ly], Kc = Kcs[sc][ly];
            convw_kernel<<<(cout * Kc + 255) / 256, 256, 0, stream>>>(
                P(sc, ly, 0), Wb, cout, cin, Kc);
            zero_kernel<<<1, 256, 0, stream>>>(sumBuf, 256);
            zero_kernel<<<1, 256, 0, stream>>>(sqBuf, 256);
            launch_gemm(Abuf, Wb, Ybuf, sumBuf, sqBuf, M, cout, Kc, stream);
            if (ly < 2)
                bn_prelu_kernel<<<2048, 256, 0, stream>>>(
                    Ybuf, sumBuf, sqBuf, P(sc, ly, 2), P(sc, ly, 3), P(sc, ly, 4),
                    Abuf, M, cout);
            else
                bn_prelu_max_kernel<<<(BS * cout + 255) / 256, 256, 0, stream>>>(
                    Ybuf, sumBuf, sqBuf, P(sc, ly, 2), P(sc, ly, 3), P(sc, ly, 4),
                    Cat, BS, K, cout, 384, coff);
        }
        coff += couts[sc][2];
    }

    // Final MLP over concat [BS, 384]
    {
        int M = BS;
        for (int ly = 0; ly < 3; ++ly) {
            int cin = cins[2][ly], cout = couts[2][ly], Kc = Kcs[2][ly];
            convw_kernel<<<(cout * Kc + 255) / 256, 256, 0, stream>>>(
                P(2, ly, 0), Wb, cout, cin, Kc);
            zero_kernel<<<1, 256, 0, stream>>>(sumBuf, 256);
            zero_kernel<<<1, 256, 0, stream>>>(sqBuf, 256);
            launch_gemm((ly == 0) ? Cat : Abuf, Wb, Ybuf, sumBuf, sqBuf,
                        M, cout, Kc, stream);
            if (ly < 2)
                bn_prelu_kernel<<<256, 256, 0, stream>>>(
                    Ybuf, sumBuf, sqBuf, P(2, ly, 2), P(2, ly, 3), P(2, ly, 4),
                    Abuf, M, cout);
            else
                bn_prelu_out_kernel<<<1024, 256, 0, stream>>>(
                    Ybuf, sumBuf, sqBuf, P(2, ly, 2), P(2, ly, 3), P(2, ly, 4),
                    (float*)d_out, M, cout);
        }
    }
}
